// DINV1_82145544503707
// MI455X (gfx1250) — matrix operand order repaired and checked by reading, not compiled, not run on hardware
//
#include <hip/hip_runtime.h>
#include <hip/hip_bf16.h>

typedef __attribute__((ext_vector_type(8)))  _Float16 v8h;
typedef __attribute__((ext_vector_type(16))) _Float16 v16h;
typedef __attribute__((ext_vector_type(8)))  float    v8f;

union AV { v16h v; v8h h[2]; };

#define BATCH 2048
#define SEQ   200
#define SP    208      // SEQ padded to 13 tiles of 16
#define DIMD  64
#define LDH   72       // LDS row stride in halfs (144B, 16B-aligned)

// ---------------------------------------------------------------------------
// Prep: convert f32 weight (K x N, row-major) -> f16 transposed (N x K)
// ---------------------------------------------------------------------------
__global__ void tconv_kernel(const float* __restrict__ W, _Float16* __restrict__ WT,
                             int K, int N) {
    int i = blockIdx.x * blockDim.x + threadIdx.x;
    if (i >= K * N) return;
    int k = i / N, n = i % N;
    WT[n * K + k] = (_Float16)W[i];
}

// ---------------------------------------------------------------------------
// Attention mega-kernel: one block (256 thr = 8 waves) per batch row.
// Produces x0h[b][192] = f16 concat(interest, q, tc)
// ---------------------------------------------------------------------------
__global__ __launch_bounds__(256) void din_attention_kernel(
    const int*   __restrict__ hist_items,
    const int*   __restrict__ maskp,
    const int*   __restrict__ target_item,
    const int*   __restrict__ target_cat,
    const float* __restrict__ item_emb,
    const float* __restrict__ cat_emb,
    const _Float16* __restrict__ aW1T,   // [64][256]
    const float* __restrict__ ab1,
    const _Float16* __restrict__ aW2T,   // [64][64]
    const float* __restrict__ ab2,
    const float* __restrict__ aW3,       // [64]
    const float* __restrict__ ab3,       // [1]
    _Float16* __restrict__ x0h)          // [B][192]
{
    __shared__ __align__(16) _Float16 qh[64];
    __shared__ __align__(16) _Float16 keysh[SP * LDH];
    __shared__ __align__(16) _Float16 hscr[8 * 16 * LDH];   // per-wave 16x64 scratch
    __shared__ float scores[SP];
    __shared__ float red[64];
    __shared__ float isum[4 * 64];

    const int b    = blockIdx.x;
    const int tid  = threadIdx.x;
    const int lane = tid & 31;
    const int wid  = tid >> 5;

    const int ti = target_item[b];
    if (tid < 64) qh[tid] = (_Float16)item_emb[(size_t)ti * 64 + tid];

    for (int i = tid; i < SP * 64; i += 256) {
        int s = i >> 6, d = i & 63;
        float v = 0.f;
        if (s < SEQ) v = item_emb[(size_t)hist_items[b * SEQ + s] * 64 + d];
        keysh[s * LDH + d] = (_Float16)v;
    }
    __syncthreads();

    const int row   = lane & 15;
    const int grp   = lane >> 4;
    const int koff  = grp * 8;    // A-fragment chunk offset
    const int kbase = grp * 8;    // B-fragment k base
    _Float16* hs = hscr + wid * 16 * LDH;

    for (int mt = wid; mt < 13; mt += 8) {
        const int srow = mt * 16 + row;
        const _Float16* kp = keysh + srow * LDH;

        // Build A fragments for all 8 K-steps of att_in = [q, k, q-k, q*k]
        AV afr[8];
        #pragma unroll
        for (int kt = 0; kt < 8; ++kt) {
            int f  = kt >> 1;
            int cb = (kt & 1) * 32 + koff;
            v8h k0 = *(const v8h*)(kp + cb);
            v8h k1 = *(const v8h*)(kp + cb + 16);
            v8h q0 = *(const v8h*)(qh + cb);
            v8h q1 = *(const v8h*)(qh + cb + 16);
            v8h a0, a1;
            if      (f == 0) { a0 = q0;      a1 = q1;      }
            else if (f == 1) { a0 = k0;      a1 = k1;      }
            else if (f == 2) { a0 = q0 - k0; a1 = q1 - k1; }
            else             { a0 = q0 * k0; a1 = q1 * k1; }
            afr[kt].h[0] = a0; afr[kt].h[1] = a1;
        }

        // Layer 1: (16x256) @ (256x64)
        v8f acc[4];
        #pragma unroll
        for (int nt = 0; nt < 4; ++nt) {
            v8f c = {};
            const _Float16* wp = aW1T + (nt * 16 + row) * 256 + kbase;
            #pragma unroll
            for (int kt = 0; kt < 8; ++kt) {
                AV bf;
                bf.h[0] = *(const v8h*)(wp + kt * 32);
                bf.h[1] = *(const v8h*)(wp + kt * 32 + 16);
                c = __builtin_amdgcn_wmma_f32_16x16x32_f16(false, afr[kt].v,
                                                           false, bf.v,
                                                           (short)0, c, false, false);
            }
            acc[nt] = c;
        }

        // bias + relu -> wave-private LDS tile (D-layout -> row-major)
        #pragma unroll
        for (int nt = 0; nt < 4; ++nt) {
            int col = nt * 16 + row;
            float bia = ab1[col];
            #pragma unroll
            for (int j = 0; j < 8; ++j) {
                float v = acc[nt][j] + bia;
                v = v > 0.f ? v : 0.f;
                hs[(grp * 8 + j) * LDH + col] = (_Float16)v;
            }
        }
        asm volatile("s_wait_dscnt 0" ::: "memory");   // same-wave LDS transpose fence
        __builtin_amdgcn_wave_barrier();

        // Layer 2: (16x64) @ (64x64)
        v8f acc2[4];
        {
            AV a2[2];
            const _Float16* hp = hs + row * LDH;
            #pragma unroll
            for (int kt = 0; kt < 2; ++kt) {
                int cb = kt * 32 + koff;
                a2[kt].h[0] = *(const v8h*)(hp + cb);
                a2[kt].h[1] = *(const v8h*)(hp + cb + 16);
            }
            #pragma unroll
            for (int nt = 0; nt < 4; ++nt) {
                v8f c = {};
                const _Float16* wp = aW2T + (nt * 16 + row) * 64 + kbase;
                #pragma unroll
                for (int kt = 0; kt < 2; ++kt) {
                    AV bf;
                    bf.h[0] = *(const v8h*)(wp + kt * 32);
                    bf.h[1] = *(const v8h*)(wp + kt * 32 + 16);
                    c = __builtin_amdgcn_wmma_f32_16x16x32_f16(false, a2[kt].v,
                                                               false, bf.v,
                                                               (short)0, c, false, false);
                }
                acc2[nt] = c;
            }
        }

        // Layer 3: bias+relu then dot with aW3, reduce across the 16-lane group
        float p[8];
        #pragma unroll
        for (int j = 0; j < 8; ++j) p[j] = 0.f;
        #pragma unroll
        for (int nt = 0; nt < 4; ++nt) {
            int col = nt * 16 + row;
            float b2 = ab2[col];
            float w3 = aW3[col];
            #pragma unroll
            for (int j = 0; j < 8; ++j) {
                float h2 = acc2[nt][j] + b2;
                h2 = h2 > 0.f ? h2 : 0.f;
                p[j] += h2 * w3;
            }
        }
        #pragma unroll
        for (int off = 1; off < 16; off <<= 1)
            #pragma unroll
            for (int j = 0; j < 8; ++j)
                p[j] += __shfl_xor(p[j], off, 32);

        if (row == 0) {
            float b3v = ab3[0];
            for (int j = 0; j < 8; ++j) {
                int s = mt * 16 + grp * 8 + j;
                float sc = p[j] + b3v;
                if (s >= SEQ)                sc = -1e9f;
                else if (maskp[b * SEQ + s] == 0) sc = -1e9f;
                scores[s] = sc;
            }
        }
    }
    __syncthreads();

    // ---- masked softmax over scores[0..SP) ----
    float v = (tid < SP) ? scores[tid] : -3.4e38f;
    for (int off = 16; off; off >>= 1) v = fmaxf(v, __shfl_xor(v, off, 32));
    if (lane == 0) red[wid] = v;
    __syncthreads();
    if (tid == 0) {
        float m = red[0];
        for (int i = 1; i < 8; ++i) m = fmaxf(m, red[i]);
        red[0] = m;
    }
    __syncthreads();
    float mx = red[0];
    float e = (tid < SP) ? __expf(scores[tid] - mx) : 0.f;
    float se = e;
    for (int off = 16; off; off >>= 1) se += __shfl_xor(se, off, 32);
    if (lane == 0) red[8 + wid] = se;
    __syncthreads();
    if (tid == 0) {
        float t = 0.f;
        for (int i = 0; i < 8; ++i) t += red[8 + i];
        red[1] = 1.f / t;
    }
    __syncthreads();
    if (tid < SP) scores[tid] = e * red[1];
    __syncthreads();

    // ---- interest = sum_s w_s * key_s ----
    {
        int d = tid & 63, g = tid >> 6;
        float a = 0.f;
        for (int s = g * 52; s < (g + 1) * 52; ++s)
            a += scores[s] * (float)keysh[s * LDH + d];
        isum[g * 64 + d] = a;
    }
    __syncthreads();
    if (tid < 64) {
        float inter = isum[tid] + isum[64 + tid] + isum[128 + tid] + isum[192 + tid];
        size_t base = (size_t)b * 192;
        x0h[base + tid]        = (_Float16)inter;
        x0h[base + 64 + tid]   = (_Float16)item_emb[(size_t)ti * 64 + tid];
        int tc = target_cat[b];
        x0h[base + 128 + tid]  = (_Float16)cat_emb[(size_t)tc * 64 + tid];
    }
}

// ---------------------------------------------------------------------------
// Generic f16 WMMA GEMM: Y[M x N] = Xh[M x K] @ W (given as WT[N x K] f16) + bias
// one wave per 16x16 output tile; K multiple of 32, M multiple of 16
// ---------------------------------------------------------------------------
__global__ __launch_bounds__(32) void gemm16_kernel(
    const _Float16* __restrict__ Xh, const _Float16* __restrict__ WT,
    const float* __restrict__ bias, float* __restrict__ Y,
    int M, int N, int K)
{
    int mt = blockIdx.x, nt = blockIdx.y;
    int lane = threadIdx.x;
    int row = lane & 15, grp = lane >> 4;

    const _Float16* xrow = Xh + (size_t)(mt * 16 + row) * K + grp * 8;
    const _Float16* wrow = WT + (size_t)(nt * 16 + row) * K + grp * 8;

    v8f acc = {};
    for (int k0 = 0; k0 < K; k0 += 32) {
        AV a, bf;
        a.h[0]  = *(const v8h*)(xrow + k0);
        a.h[1]  = *(const v8h*)(xrow + k0 + 16);
        bf.h[0] = *(const v8h*)(wrow + k0);
        bf.h[1] = *(const v8h*)(wrow + k0 + 16);
        acc = __builtin_amdgcn_wmma_f32_16x16x32_f16(false, a.v, false, bf.v,
                                                     (short)0, acc, false, false);
    }
    int col = nt * 16 + row;
    float bcol = bias[col];
    #pragma unroll
    for (int j = 0; j < 8; ++j)
        Y[(size_t)(mt * 16 + grp * 8 + j) * N + col] = acc[j] + bcol;
}

// ---------------------------------------------------------------------------
// BatchNorm stats: one block per column
// ---------------------------------------------------------------------------
__global__ __launch_bounds__(256) void bnstats_kernel(
    const float* __restrict__ Y, float* __restrict__ meanv, float* __restrict__ rstdv,
    int M, int N)
{
    int c = blockIdx.x;
    float s = 0.f, s2 = 0.f;
    for (int r = threadIdx.x; r < M; r += 256) {
        float v = Y[(size_t)r * N + c];
        s += v; s2 += v * v;
    }
    __shared__ float sh[256], sh2[256];
    sh[threadIdx.x] = s; sh2[threadIdx.x] = s2;
    __syncthreads();
    for (int off = 128; off; off >>= 1) {
        if (threadIdx.x < off) {
            sh[threadIdx.x]  += sh[threadIdx.x + off];
            sh2[threadIdx.x] += sh2[threadIdx.x + off];
        }
        __syncthreads();
    }
    if (threadIdx.x == 0) {
        float m   = sh[0] / (float)M;
        float var = sh2[0] / (float)M - m * m;
        meanv[c] = m;
        rstdv[c] = rsqrtf(var + 1e-5f);
    }
}

// ---------------------------------------------------------------------------
// BN apply + relu -> f16 for next GEMM
// ---------------------------------------------------------------------------
__global__ void bnapply_kernel(
    const float* __restrict__ Y, const float* __restrict__ meanv,
    const float* __restrict__ rstdv, const float* __restrict__ g,
    const float* __restrict__ beta, _Float16* __restrict__ Xout,
    int M, int N)
{
    int i = blockIdx.x * blockDim.x + threadIdx.x;
    if (i >= M * N) return;
    int c = i % N;
    float v = (Y[i] - meanv[c]) * rstdv[c] * g[c] + beta[c];
    v = v > 0.f ? v : 0.f;
    Xout[i] = (_Float16)v;
}

// ---------------------------------------------------------------------------
// Final 64 -> 1 projection
// ---------------------------------------------------------------------------
__global__ void finaldot_kernel(const _Float16* __restrict__ X,
                                const float* __restrict__ W3,
                                const float* __restrict__ b3,
                                float* __restrict__ out, int M)
{
    int b = blockIdx.x * blockDim.x + threadIdx.x;
    if (b >= M) return;
    float s = b3[0];
    for (int c = 0; c < 64; ++c) s += (float)X[(size_t)b * 64 + c] * W3[c];
    out[b] = s;
}

// ---------------------------------------------------------------------------
extern "C" void kernel_launch(void* const* d_in, const int* in_sizes, int n_in,
                              void* d_out, int out_size, void* d_ws, size_t ws_size,
                              hipStream_t stream) {
    (void)in_sizes; (void)n_in; (void)out_size; (void)ws_size;

    const int*   hist_items  = (const int*)d_in[0];
    const int*   maskp       = (const int*)d_in[3];
    const int*   target_item = (const int*)d_in[4];
    const int*   target_cat  = (const int*)d_in[5];
    const float* item_emb    = (const float*)d_in[6];
    const float* cat_emb     = (const float*)d_in[7];
    const float* aW1 = (const float*)d_in[8];
    const float* ab1 = (const float*)d_in[9];
    const float* aW2 = (const float*)d_in[10];
    const float* ab2 = (const float*)d_in[11];
    const float* aW3 = (const float*)d_in[12];
    const float* ab3 = (const float*)d_in[13];
    const float* W0 = (const float*)d_in[14];
    const float* b0 = (const float*)d_in[15];
    const float* g0 = (const float*)d_in[16];
    const float* be0 = (const float*)d_in[17];
    const float* W1 = (const float*)d_in[18];
    const float* b1 = (const float*)d_in[19];
    const float* g1 = (const float*)d_in[20];
    const float* be1 = (const float*)d_in[21];
    const float* W2 = (const float*)d_in[22];
    const float* b2 = (const float*)d_in[23];
    const float* g2 = (const float*)d_in[24];
    const float* be2 = (const float*)d_in[25];
    const float* W3 = (const float*)d_in[26];
    const float* b3 = (const float*)d_in[27];

    char* w = (char*)d_ws;
    auto alloc = [&](size_t bytes) {
        char* p = w;
        w += (bytes + 255) & ~(size_t)255;
        return p;
    };
    _Float16* aW1T = (_Float16*)alloc(64 * 256 * 2);
    _Float16* aW2T = (_Float16*)alloc(64 * 64 * 2);
    _Float16* W0T  = (_Float16*)alloc(256 * 192 * 2);
    _Float16* W1T  = (_Float16*)alloc(128 * 256 * 2);
    _Float16* W2T  = (_Float16*)alloc(64 * 128 * 2);
    _Float16* x0h  = (_Float16*)alloc((size_t)BATCH * 192 * 2);
    float*    y0   = (float*)alloc((size_t)BATCH * 256 * 4);
    _Float16* x1h  = (_Float16*)alloc((size_t)BATCH * 256 * 2);
    float*    y1   = (float*)alloc((size_t)BATCH * 128 * 4);
    _Float16* x2h  = (_Float16*)alloc((size_t)BATCH * 128 * 2);
    float*    y2   = (float*)alloc((size_t)BATCH * 64 * 4);
    _Float16* x3h  = (_Float16*)alloc((size_t)BATCH * 64 * 2);
    float* mean0 = (float*)alloc(256 * 4);
    float* rstd0 = (float*)alloc(256 * 4);
    float* mean1 = (float*)alloc(128 * 4);
    float* rstd1 = (float*)alloc(128 * 4);
    float* mean2 = (float*)alloc(64 * 4);
    float* rstd2 = (float*)alloc(64 * 4);

    // weight prep (f32 -> transposed f16)
    tconv_kernel<<<(256 * 64 + 255) / 256, 256, 0, stream>>>(aW1, aW1T, 256, 64);
    tconv_kernel<<<(64 * 64 + 255) / 256, 256, 0, stream>>>(aW2, aW2T, 64, 64);
    tconv_kernel<<<(192 * 256 + 255) / 256, 256, 0, stream>>>(W0, W0T, 192, 256);
    tconv_kernel<<<(256 * 128 + 255) / 256, 256, 0, stream>>>(W1, W1T, 256, 128);
    tconv_kernel<<<(128 * 64 + 255) / 256, 256, 0, stream>>>(W2, W2T, 128, 64);

    // attention -> x0h = f16 [interest | q | tc]
    din_attention_kernel<<<BATCH, 256, 0, stream>>>(
        hist_items, maskp, target_item, target_cat, item_emb, cat_emb,
        aW1T, ab1, aW2T, ab2, aW3, ab3, x0h);

    // MLP head with BN
    gemm16_kernel<<<dim3(BATCH / 16, 256 / 16), 32, 0, stream>>>(x0h, W0T, b0, y0, BATCH, 256, 192);
    bnstats_kernel<<<256, 256, 0, stream>>>(y0, mean0, rstd0, BATCH, 256);
    bnapply_kernel<<<(BATCH * 256 + 255) / 256, 256, 0, stream>>>(y0, mean0, rstd0, g0, be0, x1h, BATCH, 256);

    gemm16_kernel<<<dim3(BATCH / 16, 128 / 16), 32, 0, stream>>>(x1h, W1T, b1, y1, BATCH, 128, 256);
    bnstats_kernel<<<128, 256, 0, stream>>>(y1, mean1, rstd1, BATCH, 128);
    bnapply_kernel<<<(BATCH * 128 + 255) / 256, 256, 0, stream>>>(y1, mean1, rstd1, g1, be1, x2h, BATCH, 128);

    gemm16_kernel<<<dim3(BATCH / 16, 64 / 16), 32, 0, stream>>>(x2h, W2T, b2, y2, BATCH, 64, 128);
    bnstats_kernel<<<64, 256, 0, stream>>>(y2, mean2, rstd2, BATCH, 64);
    bnapply_kernel<<<(BATCH * 64 + 255) / 256, 256, 0, stream>>>(y2, mean2, rstd2, g2, be2, x3h, BATCH, 64);

    finaldot_kernel<<<(BATCH + 255) / 256, 256, 0, stream>>>(x3h, W3, b3, (float*)d_out, BATCH);
}
